// FeedBack_66941360276243
// MI455X (gfx1250) — matrix operand order repaired and checked by reading, not compiled, not run on hardware
//
#include <hip/hip_runtime.h>

// ---------------------------------------------------------------------------
// FeedBack LSTM forecaster for MI455X (gfx1250, wave32, WMMA)
//   BATCH=4096, UNITS=1024, SEQ_LEN=5, OUT_STEPS=200, NUM_FEATURES=1
//   - RT (recurrent weights, bf16, transposed) stays L2-resident (8MB/192MB).
//   - Per step: bf16 WMMA GEMM (fp32 acc) fused with gate activations,
//     fp32 cell-state update, bf16 h write (ping-pong buffers).
//   - Double-buffered async global->LDS staging (ASYNCcnt) hides fetch latency
//     behind the WMMA pipeline.
// ---------------------------------------------------------------------------

typedef __attribute__((ext_vector_type(16))) __bf16 v16bf;
typedef __attribute__((ext_vector_type(8)))  float  v8f;
typedef int vi4 __attribute__((vector_size(16)));                 // int4 (gcc vec)
typedef __attribute__((address_space(3))) vi4 lds_vi4;            // LDS int4

#define BATCH   4096
#define UNITS   1024
#define NGATE   4096      // 4 * UNITS
#define TOUT    200

#define BM      128       // batch rows per workgroup
#define BU      32        // units per workgroup (x4 gates)
#define BK      64        // K slab staged in LDS

union FragBF {
    uint4  q[2];
    v16bf  v;
};

__device__ __forceinline__ float dev_sigmoid(float z) {
    return 1.0f / (1.0f + __expf(-z));
}
__device__ __forceinline__ float dev_tanh(float z) {
    return 1.0f - 2.0f / (__expf(2.0f * z) + 1.0f);
}

// ---- async global->LDS staging (CDNA5: GLOBAL_LOAD_ASYNC_TO_LDS_B128) ------
#if __has_builtin(__builtin_amdgcn_global_load_async_to_lds_b128)
#define HAS_ASYNC_LDS 1
#else
#define HAS_ASYNC_LDS 0
#endif

__device__ __forceinline__ void stage16(const void* g, void* l) {
#if HAS_ASYNC_LDS
    __builtin_amdgcn_global_load_async_to_lds_b128(
        (vi4*)g, (lds_vi4*)l, /*offset=*/0, /*cpol=*/0);
#else
    *(uint4*)l = *(const uint4*)g;
#endif
}

// wait until at most N async ops outstanding (slab granularity = 8 per wave)
__device__ __forceinline__ void stage_wait_le8() {
#if HAS_ASYNC_LDS
    asm volatile("s_wait_asynccnt 0x8" ::: "memory");
#endif
}
__device__ __forceinline__ void stage_wait_le0() {
#if HAS_ASYNC_LDS
    asm volatile("s_wait_asynccnt 0x0" ::: "memory");
#endif
}

// per-thread staging descriptor (4 A slots + 4 B slots, 16B each)
struct StageCtx {
    const __bf16* gA[4];
    const __bf16* gB[4];
    int           lA[4];     // element offsets within an As buffer
    int           lB[4];     // element offsets within a  Bs buffer
};

__device__ __forceinline__ void stage_slab(const StageCtx& s,
                                           __bf16* AsBase, __bf16* BsBase,
                                           int koff) {
    #pragma unroll
    for (int it = 0; it < 4; ++it)
        stage16(s.gA[it] + koff, AsBase + s.lA[it]);
    #pragma unroll
    for (int it = 0; it < 4; ++it)
        stage16(s.gB[it] + koff, BsBase + s.lB[it]);
}

// ---------------------------------------------------------------------------
// One-time prep: RT[n][k] = (bf16) R[k][n]   (R is [1024][4096] row-major)
// ---------------------------------------------------------------------------
__global__ __launch_bounds__(256) void prep_rt_kernel(const float* __restrict__ R,
                                                      __bf16* __restrict__ RT) {
    unsigned i = blockIdx.x * 256u + threadIdx.x;   // i = n*1024 + k, 4M total
    unsigned n = i >> 10;
    unsigned k = i & 1023u;
    RT[i] = (__bf16)R[(size_t)k * NGATE + n];
}

// ---------------------------------------------------------------------------
// Zero initial state h (bf16) and c (f32)
// ---------------------------------------------------------------------------
__global__ __launch_bounds__(256) void zero_state_kernel(__bf16* __restrict__ h,
                                                         float* __restrict__ c) {
    unsigned i = blockIdx.x * 256u + threadIdx.x;
    h[i] = (__bf16)0.0f;
    c[i] = 0.0f;
}

// ---------------------------------------------------------------------------
// Fused LSTM step: z = h_in @ R + x*K + bias ; gates ; update c ; write h_out
// Grid: (BATCH/BM, UNITS/BU) = (32, 32), block = 256 threads = 8 waves.
// Waves laid out as 4 m-groups x 2 u-groups:
//   wave owns 32 rows (2 A-frags) x 16 units x 4 gates (4 B-frags, 8 acc).
// LDS slabs are double-buffered; slab i+1 streams in via ASYNCcnt while the
// 16 WMMAs of slab i execute.
// ---------------------------------------------------------------------------
__global__ __launch_bounds__(256) void lstm_step_kernel(
        const __bf16* __restrict__ h_in,     // [4096][1024] bf16
        __bf16*       __restrict__ h_out,    // [4096][1024] bf16
        float*        __restrict__ c,        // [4096][1024] f32, in place
        const __bf16* __restrict__ RT,       // [4096][1024] bf16 (R transposed)
        const float*  __restrict__ Kvec,     // kernel row  [4096] f32
        const float*  __restrict__ bias,     // [4096] f32
        const float*  __restrict__ x,        // per-batch scalar input
        int xstride)
{
    __shared__ __attribute__((aligned(16))) __bf16 As[2][BM][BK];       // 32 KB
    __shared__ __attribute__((aligned(16))) __bf16 Bs[2][4][BU][BK];    // 32 KB

    const int tid   = threadIdx.x;
    const int wave  = tid >> 5;
    const int lane  = tid & 31;
    const int half  = lane >> 4;      // 0: lanes 0-15, 1: lanes 16-31
    const int nlane = lane & 15;

    const int mg = wave & 3;          // m-group: rows mg*32 .. mg*32+31
    const int ug = wave >> 2;         // u-group: units ug*16 .. ug*16+15

    const int m0 = blockIdx.x * BM;   // batch tile origin
    const int u0 = blockIdx.y * BU;   // unit tile origin

    // ---- precompute per-thread staging addresses (branch-free, loop-invariant)
    StageCtx sc;
    #pragma unroll
    for (int it = 0; it < 4; ++it) {
        int slot = tid + it * 256;              // 1024 slots
        // A: 128 rows x 8 chunks of 8 bf16
        int rA  = slot >> 3;
        int chA = (slot & 7) * 8;
        sc.gA[it] = &h_in[(size_t)(m0 + rA) * UNITS + chA];
        sc.lA[it] = rA * BK + chA;
        // B: 4 gates x 32 rows x 8 chunks
        int g   = slot >> 8;
        int rem = slot & 255;
        int rB  = rem >> 3;
        int chB = (rem & 7) * 8;
        sc.gB[it] = &RT[(size_t)(g * UNITS + u0 + rB) * UNITS + chB];
        sc.lB[it] = (g * BU + rB) * BK + chB;
    }

    v8f acc[2][4];                    // [m-subtile][gate]
    #pragma unroll
    for (int mi = 0; mi < 2; ++mi)
        #pragma unroll
        for (int g = 0; g < 4; ++g)
            acc[mi][g] = (v8f){0.f,0.f,0.f,0.f,0.f,0.f,0.f,0.f};

    const int aRow0 = mg * 32 + nlane;        // A-frag rows (within As buffer)
    const int aRow1 = aRow0 + 16;
    const int bRow  = ug * 16 + nlane;        // B-frag row (within each Bs[g])

    // ---- prologue: stage slab 0 into buffer 0
    stage_slab(sc, &As[0][0][0], &Bs[0][0][0][0], 0);

    int pb = 0;
    for (int k0 = 0; k0 < UNITS; k0 += BK, pb ^= 1) {
        if (k0 + BK < UNITS) {
            // issue slab k0+BK into the other buffer, then wait for slab k0
            stage_slab(sc, &As[pb ^ 1][0][0], &Bs[pb ^ 1][0][0][0], k0 + BK);
            stage_wait_le8();
        } else {
            stage_wait_le0();
        }
        __syncthreads();              // slab k0 visible to all waves

        // ---- 2 WMMA K-steps per slab; batch frag loads, then 8 WMMAs
        #pragma unroll
        for (int kk = 0; kk < BK; kk += 32) {
            FragBF a[2];
            // A 16x32 bf16 layout: lanes<16 hold K 0-7 & 16-23; lanes>=16: 8-15 & 24-31
            a[0].q[0] = *(const uint4*)&As[pb][aRow0][kk + half * 8];
            a[0].q[1] = *(const uint4*)&As[pb][aRow0][kk + 16 + half * 8];
            a[1].q[0] = *(const uint4*)&As[pb][aRow1][kk + half * 8];
            a[1].q[1] = *(const uint4*)&As[pb][aRow1][kk + 16 + half * 8];
            FragBF b[4];
            #pragma unroll
            for (int g = 0; g < 4; ++g) {
                // B 32x16 layout: lanes<16 hold K 0-7 & 16-23; lanes>=16: 8-15 & 24-31
                b[g].q[0] = *(const uint4*)&Bs[pb][g][bRow][kk + half * 8];
                b[g].q[1] = *(const uint4*)&Bs[pb][g][bRow][kk + 16 + half * 8];
            }
            #pragma unroll
            for (int mi = 0; mi < 2; ++mi)
                #pragma unroll
                for (int g = 0; g < 4; ++g)
                    acc[mi][g] = __builtin_amdgcn_wmma_f32_16x16x32_bf16(
                        false, a[mi].v, false, b[g].v, (short)0, acc[mi][g],
                        false, false);
        }
        __syncthreads();              // buffer pb free for restage next iter
    }

    // ---- fused epilogue: z = acc + x*K + bias -> gates -> c, h
    // C/D layout: element r of acc @ lane -> M = half*8 + r, N = nlane
    const int u = u0 + ug * 16 + nlane;
    float kv[4], bv[4];
    #pragma unroll
    for (int g = 0; g < 4; ++g) {
        kv[g] = Kvec[g * UNITS + u];
        bv[g] = bias[g * UNITS + u];
    }
    #pragma unroll
    for (int mi = 0; mi < 2; ++mi) {
        const int m_base = m0 + mg * 32 + mi * 16 + half * 8;
        #pragma unroll
        for (int r = 0; r < 8; ++r) {
            const int b = m_base + r;
            const float xb = x[(size_t)b * xstride];
            const float zi = acc[mi][0][r] + xb * kv[0] + bv[0];
            const float zf = acc[mi][1][r] + xb * kv[1] + bv[1];
            const float zg = acc[mi][2][r] + xb * kv[2] + bv[2];
            const float zo = acc[mi][3][r] + xb * kv[3] + bv[3];
            const float ig = dev_sigmoid(zi);
            const float fg = dev_sigmoid(zf);
            const float gg = dev_tanh(zg);
            const float og = dev_sigmoid(zo);
            const size_t idx = (size_t)b * UNITS + u;
            const float cn = fg * c[idx] + ig * gg;
            c[idx]     = cn;
            h_out[idx] = (__bf16)(og * dev_tanh(cn));
        }
    }
}

// ---------------------------------------------------------------------------
// Dense head: y = relu(h.w1 + b1); val = first ? y : y*w2 + b2
// Writes out[b][t] and the feedback input x_vec[b].
// One wave (32 lanes) per batch row; 8 rows per block; 512 blocks.
// ---------------------------------------------------------------------------
__global__ __launch_bounds__(256) void head_kernel(
        const __bf16* __restrict__ h,
        const float* __restrict__ w1, const float* __restrict__ b1,
        const float* __restrict__ w2, const float* __restrict__ b2,
        float* __restrict__ out, float* __restrict__ x_vec,
        int t, int first)
{
    const int wave = threadIdx.x >> 5;
    const int lane = threadIdx.x & 31;
    const int b    = blockIdx.x * 8 + wave;

    const __bf16* hr = h + (size_t)b * UNITS;
    float s = 0.0f;
    #pragma unroll 4
    for (int j = lane; j < UNITS; j += 32)
        s += (float)hr[j] * w1[j];
    #pragma unroll
    for (int off = 16; off > 0; off >>= 1)
        s += __shfl_xor(s, off, 32);

    if (lane == 0) {
        const float y   = fmaxf(s + b1[0], 0.0f);
        const float val = first ? y : (y * w2[0] + b2[0]);
        out[(size_t)b * TOUT + t] = val;
        x_vec[b] = val;
    }
}

// ---------------------------------------------------------------------------
extern "C" void kernel_launch(void* const* d_in, const int* in_sizes, int n_in,
                              void* d_out, int out_size, void* d_ws, size_t ws_size,
                              hipStream_t stream) {
    const float* inputs = (const float*)d_in[0];   // [4096][5][1]
    const float* Kmat   = (const float*)d_in[1];   // [1][4096]
    const float* R      = (const float*)d_in[2];   // [1024][4096]
    const float* bias   = (const float*)d_in[3];   // [4096]
    const float* w1     = (const float*)d_in[4];   // [1024][1]
    const float* b1     = (const float*)d_in[5];   // [1]
    const float* w2     = (const float*)d_in[6];   // [1][1]
    const float* b2     = (const float*)d_in[7];   // [1]
    float* out = (float*)d_out;                    // [4096][200][1]

    char* ws = (char*)d_ws;
    __bf16* RT = (__bf16*)(ws);                         //  8 MB
    __bf16* hA = (__bf16*)(ws + (size_t)( 8u << 20));   //  8 MB
    __bf16* hB = (__bf16*)(ws + (size_t)(16u << 20));   //  8 MB
    float*  c  = (float* )(ws + (size_t)(24u << 20));   // 16 MB
    float*  xv = (float* )(ws + (size_t)(40u << 20));   // 16 KB

    const unsigned NELEM_BLOCKS = (BATCH * UNITS) / 256;   // 16384
    prep_rt_kernel   <<<NELEM_BLOCKS, 256, 0, stream>>>(R, RT);
    zero_state_kernel<<<NELEM_BLOCKS, 256, 0, stream>>>(hA, c);

    dim3 grid(BATCH / BM, UNITS / BU);   // 32 x 32

    __bf16* cur = hA;
    __bf16* nxt = hB;

    // warmup over the 5-step input sequence: x_t = inputs[b*5 + t]
    for (int t = 0; t < 5; ++t) {
        lstm_step_kernel<<<grid, 256, 0, stream>>>(cur, nxt, c, RT, Kmat, bias,
                                                   inputs + t, 5);
        __bf16* tmp = cur; cur = nxt; nxt = tmp;
    }

    // first prediction: dense1 only (relu), fed back as next input
    head_kernel<<<BATCH / 8, 256, 0, stream>>>(cur, w1, b1, w2, b2, out, xv,
                                               /*t=*/0, /*first=*/1);

    // autoregressive decode
    for (int t = 1; t < TOUT; ++t) {
        lstm_step_kernel<<<grid, 256, 0, stream>>>(cur, nxt, c, RT, Kmat, bias,
                                                   xv, 1);
        __bf16* tmp = cur; cur = nxt; nxt = tmp;
        head_kernel<<<BATCH / 8, 256, 0, stream>>>(cur, w1, b1, w2, b2, out, xv,
                                                   t, /*first=*/0);
    }
}
